// SW_HSS3D_91130616086887
// MI455X (gfx1250) — matrix operand order repaired and checked by reading, not compiled, not run on hardware
//
#include <hip/hip_runtime.h>
#include <hip/hip_bf16.h>
#include <stdint.h>

// ---------------- compile-time Hilbert tables (WIN=8) ----------------
constexpr int hxy2d(int n, int x, int y) {
    int d = 0;
    for (int s = n / 2; s > 0; s /= 2) {
        int rx = (x & s) ? 1 : 0;
        int ry = (y & s) ? 1 : 0;
        d += s * s * ((3 * rx) ^ ry);
        if (ry == 0) {
            if (rx == 1) { x = s - 1 - x; y = s - 1 - y; }
            int t = x; x = y; y = t;
        }
    }
    return d;
}
struct HilTab { int sorted[64]; int inv[64]; };
constexpr HilTab make_tab() {
    HilTab t{};
    for (int y = 0; y < 8; ++y)
        for (int x = 0; x < 8; ++x) {
            int d = hxy2d(8, x, y);
            t.sorted[d] = y * 8 + x;   // hilbert pos -> flat pixel
            t.inv[y * 8 + x] = d;      // flat pixel  -> hilbert pos
        }
    return t;
}
__constant__ HilTab g_tab = make_tab();

// ---------------- types ----------------
typedef __attribute__((ext_vector_type(16))) __bf16 v16bf;
typedef __attribute__((ext_vector_type(8)))  float  v8f;

// sizes
#define P_PIX   32768      // 2*128*128
#define DMODEL  128
#define DIN     256
#define KDIR    4
#define NST     16
#define RDT     8
#define LSEQ    128
#define BM      256
#define NWIN    512

// load 8 consecutive f32 (16B-aligned) and convert to bf16 into fragment halves
__device__ __forceinline__ void ld8bf(v16bf& v, int o, const float* __restrict__ p) {
    float4 t0 = ((const float4*)p)[0];
    float4 t1 = ((const float4*)p)[1];
    v[o + 0] = (__bf16)t0.x; v[o + 1] = (__bf16)t0.y;
    v[o + 2] = (__bf16)t0.z; v[o + 3] = (__bf16)t0.w;
    v[o + 4] = (__bf16)t1.x; v[o + 5] = (__bf16)t1.y;
    v[o + 6] = (__bf16)t1.z; v[o + 7] = (__bf16)t1.w;
}

// (bm, k, l) -> linear window-pixel index (nw*64 + pix) into xcw
__device__ __forceinline__ int xs_base(int bm, int k, int l) {
    int leff = (k >= 2) ? (LSEQ - 1 - l) : l;
    int nw   = bm + ((leff & 1) << 8);          // odd seq -> second half batch
    int j    = leff >> 1;
    int p    = g_tab.sorted[j];
    int pix  = (k & 1) ? (((p & 7) << 3) | (p >> 3)) : p; // dir1 = transposed raster
    return nw * 64 + pix;
}

// ---------------- kernel 1: in_proj GEMM (bf16 WMMA) ----------------
// xz[p, 0:512] = x[p, 0:128] @ W^T ; split into xv / z
__global__ __launch_bounds__(128) void k_inproj(const float* __restrict__ x,
                                                const float* __restrict__ W,
                                                float* __restrict__ xv,
                                                float* __restrict__ z) {
    int wave = threadIdx.x >> 5;
    int lane = threadIdx.x & 31;
    int row  = lane & 15;
    int kh   = lane >> 4;
    int m0   = blockIdx.x * 16;
    int n0   = (blockIdx.y * 4 + wave) * 16;

    v8f acc = {};
    const float* arow = x + (size_t)(m0 + row) * DMODEL;
    const float* brow = W + (size_t)(n0 + row) * DMODEL;   // W is (512,128): B[k][n]=W[n][k]
    for (int kk = 0; kk < DMODEL; kk += 32) {
        v16bf a, b;
        ld8bf(a, 0, arow + kk + kh * 8);
        ld8bf(a, 8, arow + kk + kh * 8 + 16);
        ld8bf(b, 0, brow + kk + kh * 8);
        ld8bf(b, 8, brow + kk + kh * 8 + 16);
        acc = __builtin_amdgcn_wmma_f32_16x16x32_bf16(false, a, false, b,
                                                      (short)0, acc, false, false);
    }
#pragma unroll
    for (int i = 0; i < 8; ++i) {
        int m = m0 + kh * 8 + i;
        int n = n0 + row;
        float v = acc[i];
        if (n < DIN) xv[(size_t)m * DIN + n] = v;
        else          z[(size_t)m * DIN + (n - DIN)] = v;
    }
}

// ---------------- kernel 2: depthwise 3x3 conv + SiLU -> window-major ----------------
__global__ __launch_bounds__(256) void k_conv(const float* __restrict__ xv,
                                              const float* __restrict__ cw,
                                              const float* __restrict__ cb,
                                              float* __restrict__ xcw) {
    int c = threadIdx.x;
    int p = blockIdx.x;
    int b  = p >> 14;
    int y  = (p >> 7) & 127;
    int xc = p & 127;
    float s = cb[c];
#pragma unroll
    for (int dy = -1; dy <= 1; ++dy) {
#pragma unroll
        for (int dx = -1; dx <= 1; ++dx) {
            int yy = y + dy, xx = xc + dx;
            if (yy >= 0 && yy < 128 && xx >= 0 && xx < 128) {
                s += xv[(((size_t)b * 128 + yy) * 128 + xx) * DIN + c] *
                     cw[c * 9 + (dy + 1) * 3 + (dx + 1)];
            }
        }
    }
    s = s / (1.f + __expf(-s));          // SiLU
    int nw  = ((b * 16 + (y >> 3)) * 16 + (xc >> 3));
    int pix = ((y & 7) * 8 + (xc & 7));
    xcw[((size_t)nw * 64 + pix) * DIN + c] = s;
}

// ---------------- kernel 3: zero accumulator ----------------
__global__ void k_zero(float* __restrict__ p, int n) {
    int i = blockIdx.x * 256 + threadIdx.x;
    if (i < n) p[i] = 0.f;
}

// ---------------- kernel 4: x_dbl = x_proj_w @ xs  (bf16 WMMA) ----------------
// per (bm,k): M=40 (pad 48), N=L=128, K=256
__global__ __launch_bounds__(128) void k_xdbl(const float* __restrict__ xcw,
                                              const float* __restrict__ xp,
                                              float* __restrict__ xdbl) {
    int wave = threadIdx.x >> 5;
    int lane = threadIdx.x & 31;
    int row  = lane & 15;
    int kh   = lane >> 4;
    int bmk  = blockIdx.x;           // bm*4 + k
    int bm   = bmk >> 2;
    int kd   = bmk & 3;
    int t    = blockIdx.y * 4 + wave;   // 0..23
    int m0   = (t >> 3) * 16;           // 0,16,32
    int l0   = (t & 7) * 16;

    int mrow = m0 + row;
    int bbase = xs_base(bm, kd, l0 + row);           // B column = seq position
    const float* ap = xp + ((size_t)kd * 40 + mrow) * DIN;
    const float* bp = xcw + (size_t)bbase * DIN;

    v8f acc = {};
    for (int kk = 0; kk < DIN; kk += 32) {
        v16bf a, b;
        if (mrow < 40) {
            ld8bf(a, 0, ap + kk + kh * 8);
            ld8bf(a, 8, ap + kk + kh * 8 + 16);
        } else {
#pragma unroll
            for (int j = 0; j < 16; ++j) a[j] = (__bf16)0.f;
        }
        ld8bf(b, 0, bp + kk + kh * 8);
        ld8bf(b, 8, bp + kk + kh * 8 + 16);
        acc = __builtin_amdgcn_wmma_f32_16x16x32_bf16(false, a, false, b,
                                                      (short)0, acc, false, false);
    }
#pragma unroll
    for (int i = 0; i < 8; ++i) {
        int m = m0 + kh * 8 + i;
        if (m < 40)
            xdbl[((size_t)bmk * 40 + m) * LSEQ + (l0 + row)] = acc[i];
    }
}

// ---------------- kernel 5: selective scan + D*u, atomic merge of 4 dirs ----------------
// x_dbl slab (20 KB) staged into LDS via gfx1250 async LDS DMA (ASYNCcnt-tracked).
__global__ __launch_bounds__(256) void k_scan(const float* __restrict__ xcw,
                                              const float* __restrict__ xdbl,
                                              const float* __restrict__ dtw,
                                              const float* __restrict__ dtb,
                                              const float* __restrict__ A_logs,
                                              const float* __restrict__ Ds,
                                              float* __restrict__ y_acc) {
    __shared__ float sdl[40 * LSEQ];     // x_dbl slab for this (bm,k): 20 KB
    int bmk = blockIdx.x;
    int bm  = bmk >> 2;
    int kd  = bmk & 3;
    int d   = threadIdx.x;

    // async global -> LDS copy: 5120 floats = 256 threads * 5 * float4
    {
        const float* src = xdbl + (size_t)bmk * 40 * LSEQ;
#pragma unroll
        for (int it = 0; it < 5; ++it) {
            int idx = (it * 256 + d) * 4;                     // float index, 16B chunks
            unsigned ldsa = (unsigned)(uintptr_t)(&sdl[idx]); // LDS byte offset (low 32b)
            const float* g = src + idx;
            asm volatile("global_load_async_to_lds_b128 %0, %1, off"
                         :: "v"(ldsa), "v"(g) : "memory");
        }
    }

    // overlap: per-thread constants while the DMA is in flight
    int kdd = kd * DIN + d;
    float w[RDT];
#pragma unroll
    for (int r = 0; r < RDT; ++r) w[r] = dtw[kdd * RDT + r];
    float bias = dtb[kdd];
    float Dv   = Ds[kdd];
    float Arow[NST];
#pragma unroll
    for (int n = 0; n < NST; ++n) Arow[n] = -__expf(A_logs[kdd * NST + n]);

    float h[NST];
#pragma unroll
    for (int n = 0; n < NST; ++n) h[n] = 0.f;

    asm volatile("s_wait_asynccnt 0x0" ::: "memory");
    __syncthreads();

    for (int l = 0; l < LSEQ; ++l) {
        float dtv = bias;
#pragma unroll
        for (int r = 0; r < RDT; ++r) dtv += w[r] * sdl[r * LSEQ + l];
        dtv = (dtv > 20.f) ? dtv : __logf(1.f + __expf(dtv));   // softplus

        int base = xs_base(bm, kd, l);
        float u  = xcw[(size_t)base * DIN + d];
        if (l + 1 < LSEQ) {
            int nb = xs_base(bm, kd, l + 1);
            __builtin_prefetch(&xcw[(size_t)nb * DIN + d], 0, 0);  // global_prefetch_b8
        }
        float du = dtv * u;
        float y  = 0.f;
#pragma unroll
        for (int n = 0; n < NST; ++n) {
            float dA = __expf(dtv * Arow[n]);
            h[n] = h[n] * dA + du * sdl[(RDT + n) * LSEQ + l];
            y   += h[n] * sdl[(RDT + NST + n) * LSEQ + l];
        }
        atomicAdd(&y_acc[(size_t)base * DIN + d], y + Dv * u);
    }
}

// ---------------- kernel 6: LayerNorm over channels + SiLU gate ----------------
__global__ __launch_bounds__(256) void k_ln_gate(const float* __restrict__ y_acc,
                                                 const float* __restrict__ z,
                                                 const float* __restrict__ lnw,
                                                 const float* __restrict__ lnb,
                                                 float* __restrict__ ygate) {
    __shared__ float sm[256];
    int q = blockIdx.x;        // nw*64 + pix
    int c = threadIdx.x;
    float v = y_acc[(size_t)q * DIN + c];

    sm[c] = v; __syncthreads();
    for (int off = 128; off > 0; off >>= 1) {
        if (c < off) sm[c] += sm[c + off];
        __syncthreads();
    }
    float mu = sm[0] * (1.f / 256.f);
    __syncthreads();
    float dv = v - mu;
    sm[c] = dv * dv; __syncthreads();
    for (int off = 128; off > 0; off >>= 1) {
        if (c < off) sm[c] += sm[c + off];
        __syncthreads();
    }
    float var = sm[0] * (1.f / 256.f);
    float rstd = rsqrtf(var + 1e-5f);

    int nw = q >> 6, pix = q & 63;
    int b  = nw >> 8, wy = (nw >> 4) & 15, wx = nw & 15;
    int py = pix >> 3, px = pix & 7;
    int gp = ((b * 128 + wy * 8 + py) * 128 + wx * 8 + px);

    float zz = z[(size_t)gp * DIN + c];
    float g  = zz / (1.f + __expf(-zz));
    ygate[(size_t)gp * DIN + c] = (dv * rstd * lnw[c] + lnb[c]) * g;
}

// ---------------- kernel 7: out_proj GEMM (bf16 WMMA) ----------------
__global__ __launch_bounds__(128) void k_outproj(const float* __restrict__ yg,
                                                 const float* __restrict__ W,
                                                 float* __restrict__ out) {
    int wave = threadIdx.x >> 5;
    int lane = threadIdx.x & 31;
    int row  = lane & 15;
    int kh   = lane >> 4;
    int m0   = blockIdx.x * 16;
    int n0   = (blockIdx.y * 4 + wave) * 16;

    v8f acc = {};
    const float* arow = yg + (size_t)(m0 + row) * DIN;
    const float* brow = W + (size_t)(n0 + row) * DIN;   // W is (128,256): B[k][n]=W[n][k]
    for (int kk = 0; kk < DIN; kk += 32) {
        v16bf a, b;
        ld8bf(a, 0, arow + kk + kh * 8);
        ld8bf(a, 8, arow + kk + kh * 8 + 16);
        ld8bf(b, 0, brow + kk + kh * 8);
        ld8bf(b, 8, brow + kk + kh * 8 + 16);
        acc = __builtin_amdgcn_wmma_f32_16x16x32_bf16(false, a, false, b,
                                                      (short)0, acc, false, false);
    }
#pragma unroll
    for (int i = 0; i < 8; ++i) {
        int m = m0 + kh * 8 + i;
        int n = n0 + row;
        out[(size_t)m * DMODEL + n] = acc[i];
    }
}

// ---------------- host launcher ----------------
extern "C" void kernel_launch(void* const* d_in, const int* in_sizes, int n_in,
                              void* d_out, int out_size, void* d_ws, size_t ws_size,
                              hipStream_t stream) {
    (void)in_sizes; (void)n_in; (void)out_size; (void)ws_size;
    const float* x          = (const float*)d_in[0];
    const float* in_proj_w  = (const float*)d_in[1];
    const float* conv_w     = (const float*)d_in[2];
    const float* conv_b     = (const float*)d_in[3];
    const float* x_proj_w   = (const float*)d_in[4];
    const float* dt_projs_w = (const float*)d_in[5];
    const float* dt_projs_b = (const float*)d_in[6];
    const float* A_logs     = (const float*)d_in[7];
    const float* Ds         = (const float*)d_in[8];
    const float* ln_w       = (const float*)d_in[9];
    const float* ln_b       = (const float*)d_in[10];
    const float* out_proj_w = (const float*)d_in[11];
    float* out = (float*)d_out;
    float* ws  = (float*)d_ws;

    const size_t PD = (size_t)P_PIX * DIN;          // 8,388,608 floats
    float* z     = ws;
    float* xv    = ws + PD;
    float* xcw   = ws + 2 * PD;
    float* xdbl  = ws + 3 * PD;                     // 1024*40*128 = 5,242,880 floats
    float* yacc  = ws + 3 * PD + (size_t)BM * KDIR * 40 * LSEQ;
    float* ygate = xv;                              // xv is dead after conv

    k_inproj <<<dim3(P_PIX / 16, 8), 128, 0, stream>>>(x, in_proj_w, xv, z);
    k_conv   <<<P_PIX, 256, 0, stream>>>(xv, conv_w, conv_b, xcw);
    k_zero   <<<(int)(PD / 256), 256, 0, stream>>>(yacc, (int)PD);
    k_xdbl   <<<dim3(BM * KDIR, 6), 128, 0, stream>>>(xcw, x_proj_w, xdbl);
    k_scan   <<<BM * KDIR, 256, 0, stream>>>(xcw, xdbl, dt_projs_w, dt_projs_b,
                                             A_logs, Ds, yacc);
    k_ln_gate<<<P_PIX, 256, 0, stream>>>(yacc, z, ln_w, ln_b, ygate);
    k_outproj<<<dim3(P_PIX / 16, 2), 128, 0, stream>>>(ygate, out_proj_w, out);
}
